// MainLSTMNNModel_66065186947205
// MI455X (gfx1250) — matrix operand order repaired and checked by reading, not compiled, not run on hardware
//
#include <hip/hip_runtime.h>
#include <hip/hip_bf16.h>
#include <math.h>

// ---------------------------------------------------------------------------
// MI455X (gfx1250) implementation of the stacked-LSTM + attention-gate model.
// All GEMMs run on v_wmma_f32_16x16x32_f16 (fp16 in, fp32 accumulate).
// Sequential T dependence handled by stream-ordered launches (global barrier).
// ---------------------------------------------------------------------------

typedef __attribute__((ext_vector_type(16))) _Float16 v16h;
typedef __attribute__((ext_vector_type(8)))  _Float16 v8h;
typedef __attribute__((ext_vector_type(4)))  _Float16 h4;
typedef __attribute__((ext_vector_type(8)))  float    v8f;

#define BN   128   // batch
#define TN   128   // time steps
#define INR  75    // real input width
#define INP  96    // padded input width (3 x 32 K-blocks)
#define HN   512   // hidden
#define CN   60    // classes
#define LDSTR 40   // LDS row stride in f16 (80B: 16B aligned, conflict-friendly)

__device__ __forceinline__ float sigmf(float x) { return 1.0f / (1.0f + expf(-x)); }

// ---------------------------------------------------------------------------
// Generic WMMA GEMM:  C[128 x N] = A1[128 x K1] @ W1 (+ A2[128 x K2] @ W2) + bias
// W* are f16, pre-packed per (32K x 16N) tile in B-fragment lane order:
//   tile(kb,nt) at offset (nt*nKb + kb)*512; within tile index = lane*16 + e,
//   element (lane,e) -> k = kb*32 + (lane>=16 ? 8 : 0) + (e<8 ? e : e+8), n = nt*16 + lane%16
// Template: NT = N tiles per block (grid.x*NT*16 == N exactly),
//           TWO = fuse second GEMM term, ACT: 0 none, 1 relu, 2 tanh.
// Launch: block = 256 (8 waves; wave w owns rows w*16..w*16+15, M = 128 = batch)
// ---------------------------------------------------------------------------
template <int NT, bool TWO, int ACT>
__global__ __launch_bounds__(256) void gemm_wmma(
    const float* __restrict__ A1, int K1, const _Float16* __restrict__ W1,
    const float* __restrict__ A2, int K2, const _Float16* __restrict__ W2,
    const float* __restrict__ bias, float* __restrict__ Cout, int N)
{
  __shared__ __align__(16) _Float16 As[128 * LDSTR];
  const int tid  = threadIdx.x;
  const int wv   = tid >> 5;
  const int lane = tid & 31;
  const int m0   = wv << 4;
  const int nt0  = blockIdx.x * NT;
  const int lh   = (lane >> 4) & 1;   // lane half (0: lanes 0-15, 1: 16-31)
  const int ml   = lane & 15;

  v8f acc[NT] = {};

  const int npass = TWO ? 2 : 1;
  for (int pass = 0; pass < npass; ++pass) {
    const float*    A = (TWO && pass) ? A2 : A1;
    const _Float16* W = (TWO && pass) ? W2 : W1;
    const int       K = (TWO && pass) ? K2 : K1;
    const int nKb = K >> 5;
    for (int kb = 0; kb < nKb; ++kb) {
      // ---- stage A block [128 x 32] fp32 -> f16 in LDS ----
      {
        const int row  = tid >> 1;
        const int col0 = (tid & 1) << 4;
        const float4* s4 = (const float4*)(A + (size_t)row * K + (kb << 5) + col0);
        _Float16*     dst = As + row * LDSTR + col0;
        #pragma unroll
        for (int u = 0; u < 4; ++u) {
          const float4 f = s4[u];
          h4 p;
          p.x = (_Float16)f.x; p.y = (_Float16)f.y;
          p.z = (_Float16)f.z; p.w = (_Float16)f.w;
          *(h4*)(dst + 4 * u) = p;           // ds_store_b64
        }
      }
      __syncthreads();
      // ---- build A fragment per ISA 16-bit A layout ----
      // lanes 0-15: K runs [0..7] and [16..23]; lanes 16-31: [8..15] and [24..31]
      const _Float16* rp = As + (m0 + ml) * LDSTR + (lh << 3);
      const v8h lo = *(const v8h*)(rp);       // 16B-aligned ds_load_b128
      const v8h hi = *(const v8h*)(rp + 16);
      const v16h af = __builtin_shufflevector(
          lo, hi, 0, 1, 2, 3, 4, 5, 6, 7, 8, 9, 10, 11, 12, 13, 14, 15);
      // ---- NT WMMAs reuse the A fragment across the wave's N tiles ----
      #pragma unroll
      for (int j = 0; j < NT; ++j) {
        const v16h bf =
            *(const v16h*)(W + (((size_t)(nt0 + j) * nKb + kb) << 9) + (lane << 4));
        acc[j] = __builtin_amdgcn_wmma_f32_16x16x32_f16(
            false, af, false, bf, (short)0, acc[j], false, false);
      }
      if (kb + 1 < nKb)  // pull next K tile of weights (global_prefetch_b8)
        __builtin_prefetch(W + (((size_t)nt0 * nKb + kb + 1) << 9), 0, 1);
      __syncthreads();
    }
  }

  // ---- epilogue: C/D layout -> lane%16 = column, VGPR r -> row r (+8 hi half)
  #pragma unroll
  for (int j = 0; j < NT; ++j) {
    const int   n  = ((nt0 + j) << 4) + ml;
    const float bv = bias[n];
    #pragma unroll
    for (int r = 0; r < 8; ++r) {
      const int m = m0 + r + (lh << 3);
      float v = acc[j][r] + bv;
      if (ACT == 1)      v = fmaxf(v, 0.0f);
      else if (ACT == 2) v = tanhf(v);
      Cout[(size_t)m * N + n] = v;
    }
  }
}

// ---------------------------------------------------------------------------
// Weight packer: dst f16 tiles in B-fragment order.
// transpose==1: logical W[k][n] = src[n*ld + k]; else src[k*ld + n].
// ---------------------------------------------------------------------------
__global__ void pack_weight(_Float16* __restrict__ dst, const float* __restrict__ src,
                            int nKb, int nNt, int Kr, int Nr, int ld, int transpose)
{
  const size_t idx   = (size_t)blockIdx.x * blockDim.x + threadIdx.x;
  const size_t total = (size_t)nKb * nNt * 512;
  if (idx >= total) return;
  const int e    = (int)(idx & 15);
  const int lane = (int)((idx >> 4) & 31);
  const int tile = (int)(idx >> 9);
  const int kb   = tile % nKb;
  const int nt   = tile / nKb;
  const int k    = (kb << 5) + ((lane >= 16) ? 8 : 0) + ((e < 8) ? e : 8 + e);
  const int n    = (nt << 4) + (lane & 15);
  float v = 0.0f;
  if (k < Kr && n < Nr)
    v = transpose ? src[(size_t)n * ld + k] : src[(size_t)k * ld + n];
  dst[idx] = (_Float16)v;
}

// M = U_s @ fc2_w^T  (512 x 25), fp32 staging before packing
__global__ void compute_M(const float* __restrict__ Us, const float* __restrict__ fc2w,
                          float* __restrict__ M)
{
  const int idx = blockIdx.x * blockDim.x + threadIdx.x;
  if (idx >= HN * 25) return;
  const int k = idx / 25, n = idx % 25;
  float s = 0.0f;
  for (int j = 0; j < HN; ++j) s += Us[(size_t)k * HN + j] * fc2w[(size_t)n * HN + j];
  M[idx] = s;
}

// bq[n] = b_us @ fc2_w^T + fc2_b (padded to 32)
__global__ void compute_bq(const float* __restrict__ bus, const float* __restrict__ fc2w,
                           const float* __restrict__ fc2b, float* __restrict__ bq)
{
  const int n = threadIdx.x;
  if (n >= 32) return;
  float s = 0.0f;
  if (n < 25) {
    s = fc2b[n];
    for (int j = 0; j < HN; ++j) s += bus[j] * fc2w[(size_t)n * HN + j];
  }
  bq[n] = s;
}

__global__ void vec_add(float* __restrict__ d, const float* __restrict__ a,
                        const float* __restrict__ b, int n)
{
  const int i = blockIdx.x * blockDim.x + threadIdx.x;
  if (i < n) d[i] = a[i] + b[i];
}

__global__ void pad_copy(float* __restrict__ d, const float* __restrict__ s, int n, int np)
{
  const int i = threadIdx.x;
  if (i < np) d[i] = (i < n) ? s[i] : 0.0f;
}

__global__ void zero_f32(float* __restrict__ d, int n)
{
  const int i = blockIdx.x * blockDim.x + threadIdx.x;
  if (i < n) d[i] = 0.0f;
}

// inputs [B,T,75] -> Xall [T,B,96] zero-padded
__global__ void build_X(const float* __restrict__ inputs, float* __restrict__ Xall)
{
  const int idx = blockIdx.x * blockDim.x + threadIdx.x;
  if (idx >= TN * BN * INP) return;
  const int k = idx % INP;
  const int r = idx / INP;
  const int b = r % BN;
  const int t = r / BN;
  Xall[idx] = (k < INR) ? inputs[((size_t)b * TN + t) * INR + k] : 0.0f;
}

// LSTM cell updates for cell_t (blockIdx.y==0) and cell_s (blockIdx.y==1)
__global__ void cell_update_ts(const float* __restrict__ gt, float* __restrict__ ht,
                               float* __restrict__ ct, const float* __restrict__ gs,
                               float* __restrict__ hs, float* __restrict__ cs)
{
  const int idx = blockIdx.x * blockDim.x + threadIdx.x;
  if (idx >= BN * HN) return;
  const int b = idx >> 9, n = idx & 511;
  const float* g = (blockIdx.y == 0) ? gt : gs;
  float* h = (blockIdx.y == 0) ? ht : hs;
  float* c = (blockIdx.y == 0) ? ct : cs;
  const size_t base = (size_t)b * 2048 + n;
  const float gi = g[base], gf = g[base + 512], gg = g[base + 1024], go = g[base + 1536];
  const float c2 = sigmf(gf) * c[idx] + sigmf(gi) * tanhf(gg);
  c[idx] = c2;
  h[idx] = sigmf(go) * tanhf(c2);
}

// cell1 update + torch aliasing bug (hn1 *= beta carried), emit beta_out
__global__ void cell_update_1(const float* __restrict__ g, float* __restrict__ h,
                              float* __restrict__ c, const float* __restrict__ beta,
                              float* __restrict__ out_beta)
{
  const int idx = blockIdx.x * blockDim.x + threadIdx.x;
  if (idx >= BN * HN) return;
  const int b = idx >> 9, n = idx & 511;
  const size_t base = (size_t)b * 2048 + n;
  const float gi = g[base], gf = g[base + 512], gg = g[base + 1024], go = g[base + 1536];
  const float c2 = sigmf(gf) * c[idx] + sigmf(gi) * tanhf(gg);
  const float hv = sigmf(go) * tanhf(c2);
  const float bv = beta[idx];
  c[idx] = c2;
  h[idx] = hv * bv;      // gated value is carried (reference aliasing bug)
  out_beta[idx] = bv;
}

// softmax over 25, write alpha_out, build gated input Xg = x * repeat(alpha,3)
__global__ void alpha_gate(const float* __restrict__ q, const float* __restrict__ Xt,
                           float* __restrict__ Xg, float* __restrict__ out_alpha)
{
  __shared__ float sa[25];
  const int b = blockIdx.x, tid = threadIdx.x;
  if (tid < 25) sa[tid] = q[(size_t)b * 32 + tid];
  __syncthreads();
  if (tid == 0) {
    float mx = sa[0];
    for (int j = 1; j < 25; ++j) mx = fmaxf(mx, sa[j]);
    float s = 0.0f;
    for (int j = 0; j < 25; ++j) { sa[j] = expf(sa[j] - mx); s += sa[j]; }
    const float inv = 1.0f / s;
    for (int j = 0; j < 25; ++j) sa[j] *= inv;
  }
  __syncthreads();
  if (tid < 25) out_alpha[(size_t)b * 25 + tid] = sa[tid];
  if (tid < INP) {
    const float a = (tid < INR) ? sa[tid / 3] : 0.0f;
    Xg[(size_t)b * INP + tid] = Xt[(size_t)b * INP + tid] * a;
  }
}

// log_softmax over 60 classes (logits buffer padded to 64)
__global__ void logsoftmax60(const float* __restrict__ logits, float* __restrict__ out)
{
  __shared__ float s[64];
  __shared__ float red[2];
  const int b = blockIdx.x, tid = threadIdx.x;  // blockDim = 64
  s[tid] = (tid < CN) ? logits[(size_t)b * 64 + tid] : -1e30f;
  __syncthreads();
  if (tid == 0) {
    float mx = s[0];
    for (int j = 1; j < CN; ++j) mx = fmaxf(mx, s[j]);
    float sum = 0.0f;
    for (int j = 0; j < CN; ++j) sum += expf(s[j] - mx);
    red[0] = mx;
    red[1] = logf(sum);
  }
  __syncthreads();
  if (tid < CN) out[(size_t)b * CN + tid] = s[tid] - red[0] - red[1];
}

// ---------------------------------------------------------------------------
extern "C" void kernel_launch(void* const* d_in, const int* in_sizes, int n_in,
                              void* d_out, int out_size, void* d_ws, size_t ws_size,
                              hipStream_t stream)
{
  (void)in_sizes; (void)n_in; (void)out_size; (void)ws_size;

  const float* inputs = (const float*)d_in[0];
  // d_in[1] = epoch (unused)
  const float* Wih_t = (const float*)d_in[2];
  const float* Whh_t = (const float*)d_in[3];
  const float* bih_t = (const float*)d_in[4];
  const float* bhh_t = (const float*)d_in[5];
  const float* Wih_s = (const float*)d_in[6];
  const float* Whh_s = (const float*)d_in[7];
  const float* bih_s = (const float*)d_in[8];
  const float* bhh_s = (const float*)d_in[9];
  const float* Wih_1 = (const float*)d_in[10];
  const float* Whh_1 = (const float*)d_in[11];
  const float* bih_1 = (const float*)d_in[12];
  const float* bhh_1 = (const float*)d_in[13];
  const float* fc_w  = (const float*)d_in[14];
  const float* fc_b  = (const float*)d_in[15];
  const float* fct_w = (const float*)d_in[16];
  const float* fct_b = (const float*)d_in[17];
  const float* fc1_w = (const float*)d_in[18];
  const float* fc1_b = (const float*)d_in[19];
  const float* fc2_w = (const float*)d_in[20];
  const float* fc2_b = (const float*)d_in[21];
  const float* W_x_t = (const float*)d_in[22];
  const float* W_h_t = (const float*)d_in[23];
  const float* b_t   = (const float*)d_in[24];
  const float* W_x_s = (const float*)d_in[25];
  const float* W_h_s = (const float*)d_in[26];
  const float* U_s   = (const float*)d_in[27];
  const float* b_s   = (const float*)d_in[28];
  const float* b_us  = (const float*)d_in[29];

  // ---- workspace carve-out (~22 MB), 256B aligned ----
  char* ws = (char*)d_ws;
  size_t off = 0;
  auto alloc = [&](size_t bytes) -> void* {
    void* p = ws + off;
    off = (off + bytes + 255) & ~(size_t)255;
    return p;
  };
  _Float16* pkWih_t = (_Float16*)alloc((size_t)3 * 128 * 512 * 2);
  _Float16* pkWih_s = (_Float16*)alloc((size_t)3 * 128 * 512 * 2);
  _Float16* pkWih_1 = (_Float16*)alloc((size_t)3 * 128 * 512 * 2);
  _Float16* pkWhh_t = (_Float16*)alloc((size_t)16 * 128 * 512 * 2);
  _Float16* pkWhh_s = (_Float16*)alloc((size_t)16 * 128 * 512 * 2);
  _Float16* pkWhh_1 = (_Float16*)alloc((size_t)16 * 128 * 512 * 2);
  _Float16* pkFct   = (_Float16*)alloc((size_t)3 * 32 * 512 * 2);
  _Float16* pkFc1   = (_Float16*)alloc((size_t)3 * 32 * 512 * 2);
  _Float16* pkWxt   = (_Float16*)alloc((size_t)16 * 32 * 512 * 2);
  _Float16* pkWht   = (_Float16*)alloc((size_t)16 * 32 * 512 * 2);
  _Float16* pkWxs   = (_Float16*)alloc((size_t)16 * 32 * 512 * 2);
  _Float16* pkWhs   = (_Float16*)alloc((size_t)16 * 32 * 512 * 2);
  _Float16* pkM     = (_Float16*)alloc((size_t)16 * 2 * 512 * 2);
  _Float16* pkFc    = (_Float16*)alloc((size_t)16 * 4 * 512 * 2);
  float* Xall    = (float*)alloc((size_t)TN * BN * INP * 4);
  float* Xg      = (float*)alloc((size_t)BN * INP * 4);
  float* states  = (float*)alloc((size_t)6 * BN * HN * 4);
  float* h_t = states + 0 * BN * HN;
  float* c_t = states + 1 * BN * HN;
  float* h_s = states + 2 * BN * HN;
  float* c_s = states + 3 * BN * HN;
  float* h_1 = states + 4 * BN * HN;
  float* c_1 = states + 5 * BN * HN;
  float* gates_t = (float*)alloc((size_t)BN * 2048 * 4);
  float* gates_s = (float*)alloc((size_t)BN * 2048 * 4);
  float* gates_1 = (float*)alloc((size_t)BN * 2048 * 4);
  float* xt_proj = (float*)alloc((size_t)BN * HN * 4);
  float* x1_proj = (float*)alloc((size_t)BN * HN * 4);
  float* st1     = (float*)alloc((size_t)BN * HN * 4);
  float* betab   = (float*)alloc((size_t)BN * HN * 4);
  float* qbuf    = (float*)alloc((size_t)BN * 32 * 4);
  float* logitsb = (float*)alloc((size_t)BN * 64 * 4);
  float* Mtmp    = (float*)alloc((size_t)HN * 25 * 4);
  float* bcomb_t = (float*)alloc(2048 * 4);
  float* bcomb_s = (float*)alloc(2048 * 4);
  float* bcomb_1 = (float*)alloc(2048 * 4);
  float* bbeta   = (float*)alloc(512 * 4);
  float* bq      = (float*)alloc(32 * 4);
  float* fcb_pad = (float*)alloc(64 * 4);

  float* out_logp  = (float*)d_out;                        // [B, 60]
  float* out_alpha = out_logp + (size_t)BN * CN;           // [T, B, 25] flat
  float* out_beta  = out_alpha + (size_t)TN * BN * 25;     // [T, B, 512] flat

  auto pack = [&](_Float16* dst, const float* src, int nKb, int nNt,
                  int Kr, int Nr, int ld, int tr) {
    const size_t total = (size_t)nKb * nNt * 512;
    pack_weight<<<dim3((unsigned)((total + 255) / 256)), dim3(256), 0, stream>>>(
        dst, src, nKb, nNt, Kr, Nr, ld, tr);
  };

  // ---- prep: layouts, packed weights, fused biases (recomputed each call) ----
  build_X<<<dim3((TN * BN * INP + 255) / 256), dim3(256), 0, stream>>>(inputs, Xall);
  zero_f32<<<dim3((6 * BN * HN + 255) / 256), dim3(256), 0, stream>>>(states, 6 * BN * HN);
  vec_add<<<dim3(8), dim3(256), 0, stream>>>(bcomb_t, bih_t, bhh_t, 2048);
  vec_add<<<dim3(8), dim3(256), 0, stream>>>(bcomb_s, bih_s, bhh_s, 2048);
  vec_add<<<dim3(8), dim3(256), 0, stream>>>(bcomb_1, bih_1, bhh_1, 2048);
  vec_add<<<dim3(2), dim3(256), 0, stream>>>(bbeta, b_s, b_t, 512);
  pad_copy<<<dim3(1), dim3(64), 0, stream>>>(fcb_pad, fc_b, CN, 64);
  compute_M<<<dim3((HN * 25 + 255) / 256), dim3(256), 0, stream>>>(U_s, fc2_w, Mtmp);
  compute_bq<<<dim3(1), dim3(32), 0, stream>>>(b_us, fc2_w, fc2_b, bq);

  pack(pkWih_t, Wih_t, 3, 128, INR, 2048, INR, 1);
  pack(pkWih_s, Wih_s, 3, 128, INR, 2048, INR, 1);
  pack(pkWih_1, Wih_1, 3, 128, INR, 2048, INR, 1);
  pack(pkWhh_t, Whh_t, 16, 128, HN, 2048, HN, 1);
  pack(pkWhh_s, Whh_s, 16, 128, HN, 2048, HN, 1);
  pack(pkWhh_1, Whh_1, 16, 128, HN, 2048, HN, 1);
  pack(pkFct, fct_w, 3, 32, INR, HN, INR, 1);
  pack(pkFc1, fc1_w, 3, 32, INR, HN, INR, 1);
  pack(pkWxt, W_x_t, 16, 32, HN, HN, HN, 0);
  pack(pkWht, W_h_t, 16, 32, HN, HN, HN, 0);
  pack(pkWxs, W_x_s, 16, 32, HN, HN, HN, 0);
  pack(pkWhs, W_h_s, 16, 32, HN, HN, HN, 0);
  pack(pkM, Mtmp, 16, 2, HN, 25, 25, 0);
  pack(pkFc, fc_w, 16, 4, HN, CN, HN, 1);

  // ---- sequential scan over T (launch order == global barrier) ----
  for (int t = 0; t < TN; ++t) {
    const float* Xt = Xall + (size_t)t * BN * INP;
    // gates for cell_t / cell_s, and the two x-projections (all read OLD h)
    gemm_wmma<4, true, 0><<<dim3(32), dim3(256), 0, stream>>>(
        Xt, INP, pkWih_t, h_t, HN, pkWhh_t, bcomb_t, gates_t, 2048);
    gemm_wmma<4, true, 0><<<dim3(32), dim3(256), 0, stream>>>(
        Xt, INP, pkWih_s, h_s, HN, pkWhh_s, bcomb_s, gates_s, 2048);
    gemm_wmma<4, false, 0><<<dim3(8), dim3(256), 0, stream>>>(
        Xt, INP, pkFct, nullptr, 0, nullptr, fct_b, xt_proj, HN);
    gemm_wmma<4, false, 0><<<dim3(8), dim3(256), 0, stream>>>(
        Xt, INP, pkFc1, nullptr, 0, nullptr, fc1_b, x1_proj, HN);
    // beta = relu(xt_proj@W_x_t + h_prev_t@W_h_t + b_s + b_t)  (old h_t)
    gemm_wmma<4, true, 1><<<dim3(8), dim3(256), 0, stream>>>(
        xt_proj, HN, pkWxt, h_t, HN, pkWht, bbeta, betab, HN);
    // st1 = tanh(x1_proj@W_x_s + h_prev_s@W_h_s + b_s)  (old h_s)
    gemm_wmma<4, true, 2><<<dim3(8), dim3(256), 0, stream>>>(
        x1_proj, HN, pkWxs, h_s, HN, pkWhs, b_s, st1, HN);
    // now update cell_t / cell_s states
    cell_update_ts<<<dim3(256, 2), dim3(256), 0, stream>>>(
        gates_t, h_t, c_t, gates_s, h_s, c_s);
    // q = st1 @ (U_s@fc2^T) + (b_us@fc2^T + fc2_b)
    gemm_wmma<2, false, 0><<<dim3(1), dim3(256), 0, stream>>>(
        st1, HN, pkM, nullptr, 0, nullptr, bq, qbuf, 32);
    // alpha = softmax(q); Xg = x * repeat(alpha, 3); emit alpha_out
    alpha_gate<<<dim3(BN), dim3(128), 0, stream>>>(
        qbuf, Xt, Xg, out_alpha + (size_t)t * BN * 25);
    // cell_1 gates on gated input (old h_1)
    gemm_wmma<4, true, 0><<<dim3(32), dim3(256), 0, stream>>>(
        Xg, INP, pkWih_1, h_1, HN, pkWhh_1, bcomb_1, gates_1, 2048);
    // cell_1 update + h_1 *= beta (carried), emit beta_out
    cell_update_1<<<dim3(256), dim3(256), 0, stream>>>(
        gates_1, h_1, c_1, betab, out_beta + (size_t)t * BN * HN);
  }

  // ---- head: logits = h_1 @ fc^T + fc_b; log_softmax ----
  gemm_wmma<4, false, 0><<<dim3(1), dim3(256), 0, stream>>>(
      h_1, HN, pkFc, nullptr, 0, nullptr, fcb_pad, logitsb, 64);
  logsoftmax60<<<dim3(BN), dim3(64), 0, stream>>>(logitsb, out_logp);
}
